// AttentionBlock_30116310680123
// MI455X (gfx1250) — compile-verified, matrix operand order checked by reading, not run on hardware
//
#include <hip/hip_runtime.h>
#include <hip/hip_bf16.h>

typedef _Float16 f16;
typedef __attribute__((ext_vector_type(16))) _Float16 v16h;
typedef __attribute__((ext_vector_type(8)))  _Float16 v8h;
typedef __attribute__((ext_vector_type(8)))  float    v8f;

#define TDIM 1024
#define CDIM 512
#define HEADS 8
#define M_QKV 1536

__device__ __forceinline__ v8f wmma_f16(v16h a, v16h b, v8f c) {
    return __builtin_amdgcn_wmma_f32_16x16x32_f16(false, a, false, b, (short)0, c, false, false);
}
__device__ __forceinline__ v16h cat8(v8h lo, v8h hi) {
    return __builtin_shufflevector(lo, hi, 0, 1, 2, 3, 4, 5, 6, 7, 8, 9, 10, 11, 12, 13, 14, 15);
}
// fragment = two aligned 16B loads: k 8h..8h+7 then 16+8h..16+8h+7 (h = lane/16)
__device__ __forceinline__ v16h frag(const f16* row, int kc) {
    return cat8(*(const v8h*)(row + kc), *(const v8h*)(row + kc + 16));
}

// CDNA5 async global->LDS copy: 4 consecutive 16B chunks per thread.
// IOFFSET is added to BOTH the LDS (VDST) and global (VADDR) addresses, so one
// LDS-address VGPR + one global pointer cover all four copies via immediates.
__device__ __forceinline__ void async_copy4(const f16* lds_dst, const f16* gsrc) {
    unsigned l = (unsigned)(size_t)lds_dst;   // LDS aperture: low 32 bits = LDS byte offset
    asm volatile(
        "global_load_async_to_lds_b128 %0, %1, off\n\t"
        "global_load_async_to_lds_b128 %0, %1, off offset:16\n\t"
        "global_load_async_to_lds_b128 %0, %1, off offset:32\n\t"
        "global_load_async_to_lds_b128 %0, %1, off offset:48"
        :: "v"(l), "v"(gsrc) : "memory");
}
__device__ __forceinline__ void wait_async0() {
    asm volatile("s_wait_asynccnt 0x0" ::: "memory");
}

__device__ __forceinline__ float rmax16(float v) {
    v = fmaxf(v, __shfl_xor(v, 1, 16));
    v = fmaxf(v, __shfl_xor(v, 2, 16));
    v = fmaxf(v, __shfl_xor(v, 4, 16));
    v = fmaxf(v, __shfl_xor(v, 8, 16));
    return v;
}
__device__ __forceinline__ float rsum16(float v) {
    v += __shfl_xor(v, 1, 16);
    v += __shfl_xor(v, 2, 16);
    v += __shfl_xor(v, 4, 16);
    v += __shfl_xor(v, 8, 16);
    return v;
}

// ---------------- f32 -> f16 weight conversion ----------------
__global__ void k_f32_to_f16(const float* __restrict__ s, f16* __restrict__ d, int n) {
    int i = blockIdx.x * blockDim.x + threadIdx.x;
    if (i < n) d[i] = (f16)s[i];
}

// ---------------- GroupNorm -> xnT [b][t][c] (f16) ----------------
__global__ void k_groupnorm(const float* __restrict__ x, const float* __restrict__ gamma,
                            const float* __restrict__ beta, f16* __restrict__ xnT) {
    const int bg = blockIdx.x;          // b*32 + g
    const int b = bg >> 5, g = bg & 31;
    const int tid = threadIdx.x;        // 256 threads
    const float* xp = x + ((size_t)(b * CDIM + g * 16)) * TDIM;   // 16 chans * 1024 contiguous
    float s = 0.f, ss = 0.f;
    for (int i = tid; i < 16 * TDIM; i += 256) { float v = xp[i]; s += v; ss += v * v; }
    __shared__ float red0[256], red1[256];
    red0[tid] = s; red1[tid] = ss; __syncthreads();
    for (int off = 128; off > 0; off >>= 1) {
        if (tid < off) { red0[tid] += red0[tid + off]; red1[tid] += red1[tid + off]; }
        __syncthreads();
    }
    __shared__ float mv[2];
    if (tid == 0) {
        float mean = red0[0] * (1.f / (16.f * TDIM));
        float var  = red1[0] * (1.f / (16.f * TDIM)) - mean * mean;
        mv[0] = mean; mv[1] = rsqrtf(var + 1e-5f);
    }
    __syncthreads();
    const float mean = mv[0], rstd = mv[1];
    // write transposed; cl is fixed per thread so gamma/beta hoist out of loop
    const int cl = tid & 15, c = g * 16 + cl;
    const float ga = gamma[c] * rstd, be = beta[c] - mean * rstd * gamma[c];
    for (int j = tid; j < 16 * TDIM; j += 256) {
        int t = j >> 4;
        xnT[((size_t)b * TDIM + t) * CDIM + c] = (f16)(xp[cl * TDIM + t] * ga + be);
    }
}

// ---------------- LDS-free WMMA GEMM ----------------
// OUT[b] = A(MxK, row major) x Bt[b] (N=1024 rows, K contiguous) + bias
// MODE 0: scatter into qT/kT [bh][t][64] (packed b128 stores) and vH [bh][c][t]   (QKV)
// MODE 1: f32 out + residual                                                      (proj)
template <int MODE>
__global__ void k_gemm(const f16* __restrict__ A, const f16* __restrict__ Bt,
                       const float* __restrict__ bias, int K,
                       f16* __restrict__ qT, f16* __restrict__ kT, f16* __restrict__ vH,
                       const float* __restrict__ resid, float* __restrict__ outF) {
    const int bidx = blockIdx.z;
    const int nb = blockIdx.x * 128, mb = blockIdx.y * 128;
    const int tid = threadIdx.x, lane = tid & 31, l16 = lane & 15, lh = lane >> 4;
    const int wv = tid >> 5, wm = wv >> 2, wn = wv & 3;     // wave grid 2(M) x 4(N)
    const int kc = lh * 8;
    const f16* Bp = Bt + (size_t)bidx * TDIM * K;

    v8f acc[4][2];
    v8f zf = {};
#pragma unroll
    for (int mt = 0; mt < 4; ++mt)
#pragma unroll
        for (int nt = 0; nt < 2; ++nt) acc[mt][nt] = zf;

    const f16* arow[4];
    const f16* brow[2];
#pragma unroll
    for (int mt = 0; mt < 4; ++mt) arow[mt] = A + (size_t)(mb + wm * 64 + mt * 16 + l16) * K;
#pragma unroll
    for (int nt = 0; nt < 2; ++nt) brow[nt] = Bp + (size_t)(nb + wn * 32 + nt * 16 + l16) * K;

    for (int k0 = 0; k0 < K; k0 += 32) {
        v16h af[4], bf[2];
#pragma unroll
        for (int mt = 0; mt < 4; ++mt) af[mt] = frag(arow[mt] + k0, kc);
#pragma unroll
        for (int nt = 0; nt < 2; ++nt) bf[nt] = frag(brow[nt] + k0, kc);
#pragma unroll
        for (int mt = 0; mt < 4; ++mt)
#pragma unroll
            for (int nt = 0; nt < 2; ++nt)
                acc[mt][nt] = wmma_f16(af[mt], bf[nt], acc[mt][nt]);
    }

#pragma unroll
    for (int mt = 0; mt < 4; ++mt) {
        const int row_base = mb + wm * 64 + mt * 16;            // 16-aligned tile base
        const v8f bv = *(const v8f*)(bias + row_base + lh * 8); // lane's 8 row biases
        if (MODE == 0) {
            // q/k/v partitions are 64-aligned -> uniform per tile
            const int hh = row_base / 192;
            const int rem = row_base - hh * 192;
            const int part = rem >> 6;
            const int cl0 = (rem & 63) + lh * 8;                // + r = lane's head channel
            const size_t bh = (size_t)bidx * HEADS + hh;
#pragma unroll
            for (int nt = 0; nt < 2; ++nt) {
                const int col = nb + wn * 32 + nt * 16 + l16;   // token t
                v8h pk;
#pragma unroll
                for (int r = 0; r < 8; ++r) pk[r] = (f16)(acc[mt][nt][r] + bv[r]);
                if (part == 0)
                    *(v8h*)(qT + (bh * TDIM + col) * 64 + cl0) = pk;   // 8 contiguous channels
                else if (part == 1)
                    *(v8h*)(kT + (bh * TDIM + col) * 64 + cl0) = pk;
                else {
#pragma unroll
                    for (int r = 0; r < 8; ++r)
                        vH[(bh * 64 + cl0 + r) * TDIM + col] = pk[r];  // lane-contiguous in col
                }
            }
        } else {
            float* of = outF + (size_t)bidx * CDIM * TDIM;
            const float* rp = resid + (size_t)bidx * CDIM * TDIM;
#pragma unroll
            for (int nt = 0; nt < 2; ++nt) {
                const int col = nb + wn * 32 + nt * 16 + l16;
#pragma unroll
                for (int r = 0; r < 8; ++r) {
                    const size_t idx = (size_t)(row_base + r + lh * 8) * TDIM + col;
                    of[idx] = rp[idx] + acc[mt][nt][r] + bv[r];
                }
            }
        }
    }
}

// ---------------- Flash attention ----------------
// 4 waves/block share (b,h): K/V chunks staged once per block into LDS via
// global_load_async_to_lds_b128 (ASYNCcnt), fragments read back as ds_load_b128.
__global__ void k_attention(const f16* __restrict__ qT, const f16* __restrict__ kT,
                            const f16* __restrict__ vH, f16* __restrict__ hT) {
    const int bh = blockIdx.y;          // b*8 + h
    const int b = bh >> 3, h = bh & 7;
    const int tid = threadIdx.x, lane = tid & 31, l16 = lane & 15, lh = lane >> 4;
    const int wv = tid >> 5;            // 4 waves / block
    const int t0 = blockIdx.x * 64 + wv * 16;
    const int kc = lh * 8;

    const f16* qrow = qT + ((size_t)bh * TDIM + t0 + l16) * 64;

    __shared__ f16 kbuf[64 * 64];       // K chunk: [s_local][c]
    __shared__ f16 vbuf[64 * 64];       // V chunk: [c_local][s_local]
    __shared__ f16 pbuf[4 * 16 * 72];   // per-wave 16x64 P tile, row stride 72 halves

    // fixed per-thread staging addresses: row = tid>>1, 64B column span = (tid&1)*32 halves
    const int srow = tid >> 1, soff = (tid & 1) * 32;
    const f16* kg = kT + (size_t)bh * TDIM * 64 + (size_t)srow * 64 + soff;   // +4096 halves/iter
    const f16* vg = vH + (size_t)bh * 64 * TDIM + (size_t)srow * TDIM + soff; // +64 halves/iter
    f16* kl = &kbuf[srow * 64 + soff];
    f16* vl = &vbuf[srow * 64 + soff];

    // q fragments (reused across all keys)
    const v16h aq0 = frag(qrow, kc);
    const v16h aq1 = frag(qrow + 32, kc);

    v8f zf = {};
    v8f out[4];
#pragma unroll
    for (int nt = 0; nt < 4; ++nt) out[nt] = zf;
    float mrow[8], lrow[8];
#pragma unroll
    for (int r = 0; r < 8; ++r) { mrow[r] = -3.0e38f; lrow[r] = 0.f; }

    for (int ci = 0; ci < 16; ++ci) {
        // ---- async-stage K (8KB) and V (8KB) chunks; 64B per thread per matrix ----
        async_copy4(kl, kg);
        async_copy4(vl, vg);
        kg += 64 * 64;      // next 64-key chunk of kT rows
        vg += 64;           // next 64 columns of vH rows
        wait_async0();
        __syncthreads();

        v8f sc[4];
#pragma unroll
        for (int st = 0; st < 4; ++st) {            // four 16-wide score tiles
            const f16* krow = &kbuf[(st * 16 + l16) * 64];
            v8f z = zf;
            z = wmma_f16(aq0, frag(krow, kc), z);
            z = wmma_f16(aq1, frag(krow + 32, kc), z);
            sc[st] = z * 0.125f;                    // scale^2 = 1/sqrt(64)
        }

        // online softmax (row = r + 8*lh, uniform across each 16-lane group)
        float fac[8];
#pragma unroll
        for (int r = 0; r < 8; ++r) {
            float tm = fmaxf(fmaxf(sc[0][r], sc[1][r]), fmaxf(sc[2][r], sc[3][r]));
            tm = rmax16(tm);
            const float mnew = fmaxf(mrow[r], tm);
            fac[r] = __expf(mrow[r] - mnew);
            mrow[r] = mnew;
            float ps = 0.f;
#pragma unroll
            for (int st = 0; st < 4; ++st) { sc[st][r] = __expf(sc[st][r] - mnew); ps += sc[st][r]; }
            lrow[r] = lrow[r] * fac[r] + rsum16(ps);
        }
#pragma unroll
        for (int nt = 0; nt < 4; ++nt)
#pragma unroll
            for (int r = 0; r < 8; ++r) out[nt][r] *= fac[r];

        // P tile: D-layout regs -> LDS -> A-layout fragments
#pragma unroll
        for (int st = 0; st < 4; ++st)
#pragma unroll
            for (int r = 0; r < 8; ++r)
                pbuf[wv * 1152 + (r + 8 * lh) * 72 + st * 16 + l16] = (f16)sc[st][r];
        __syncthreads();
        const f16* prow = &pbuf[wv * 1152 + l16 * 72];
        const v16h ap0 = frag(prow, kc);
        const v16h ap1 = frag(prow + 32, kc);

        // out += P (16x64) @ V^T (64x64)
#pragma unroll
        for (int nt = 0; nt < 4; ++nt) {
            const f16* vrow = &vbuf[(nt * 16 + l16) * 64];
            out[nt] = wmma_f16(ap0, frag(vrow, kc), out[nt]);
            out[nt] = wmma_f16(ap1, frag(vrow + 32, kc), out[nt]);
        }
        __syncthreads();   // protect kbuf/vbuf/pbuf before next chunk's staging
    }

    // normalize; write hT [b][t][c] (lane-contiguous along c)
#pragma unroll
    for (int nt = 0; nt < 4; ++nt) {
        const int c = h * 64 + nt * 16 + l16;
#pragma unroll
        for (int r = 0; r < 8; ++r) {
            const int t = t0 + r + 8 * lh;
            hT[((size_t)b * TDIM + t) * CDIM + c] = (f16)(out[nt][r] / lrow[r]);
        }
    }
}

extern "C" void kernel_launch(void* const* d_in, const int* in_sizes, int n_in,
                              void* d_out, int out_size, void* d_ws, size_t ws_size,
                              hipStream_t stream) {
    (void)in_sizes; (void)n_in; (void)out_size; (void)ws_size;
    const float* x      = (const float*)d_in[0];
    const float* gamma  = (const float*)d_in[1];
    const float* beta   = (const float*)d_in[2];
    const float* w_qkv  = (const float*)d_in[3];
    const float* b_qkv  = (const float*)d_in[4];
    const float* w_proj = (const float*)d_in[5];
    const float* b_proj = (const float*)d_in[6];

    char* ws = (char*)d_ws;
    size_t off = 0;
    auto carve = [&](size_t bytes) { void* p = ws + off; off += (bytes + 255) & ~(size_t)255; return p; };
    f16* wqkv_h  = (f16*)carve((size_t)M_QKV * CDIM * 2);
    f16* wproj_h = (f16*)carve((size_t)CDIM * CDIM * 2);
    f16* xnT     = (f16*)carve((size_t)8 * TDIM * CDIM * 2);
    f16* qT      = (f16*)carve((size_t)64 * TDIM * 64 * 2);
    f16* kT      = (f16*)carve((size_t)64 * TDIM * 64 * 2);
    f16* vH      = (f16*)carve((size_t)64 * 64 * TDIM * 2);
    f16* hT      = (f16*)carve((size_t)8 * TDIM * CDIM * 2);

    k_f32_to_f16<<<(M_QKV * CDIM + 255) / 256, 256, 0, stream>>>(w_qkv, wqkv_h, M_QKV * CDIM);
    k_f32_to_f16<<<(CDIM * CDIM + 255) / 256, 256, 0, stream>>>(w_proj, wproj_h, CDIM * CDIM);

    k_groupnorm<<<8 * 32, 256, 0, stream>>>(x, gamma, beta, xnT);

    dim3 g1(TDIM / 128, M_QKV / 128, 8);
    k_gemm<0><<<g1, 256, 0, stream>>>(wqkv_h, xnT, b_qkv, CDIM, qT, kT, vH, nullptr, nullptr);

    dim3 g2(TDIM / 64, 8 * HEADS);
    k_attention<<<g2, 128, 0, stream>>>(qT, kT, vH, hT);

    dim3 g3(TDIM / 128, CDIM / 128, 8);
    k_gemm<1><<<g3, 256, 0, stream>>>(wproj_h, hT, b_proj, CDIM, nullptr, nullptr, nullptr,
                                      x, (float*)d_out);
}
